// FrequencyAwareGCFM_56590489092719
// MI455X (gfx1250) — matrix operand order repaired and checked by reading, not compiled, not run on hardware
//
#include <hip/hip_runtime.h>

#define HWD (128*128)

typedef __attribute__((ext_vector_type(16))) __bf16        v16bf;
typedef __attribute__((ext_vector_type(8)))  float         v8f;
typedef __attribute__((ext_vector_type(8)))  unsigned int  v8u;

__device__ __forceinline__ unsigned short f2bf(float f){
  unsigned int u = __builtin_bit_cast(unsigned int, f);
  u = (u + 0x7FFFu + ((u >> 16) & 1u)) >> 16;
  return (unsigned short)u;
}
__device__ __forceinline__ float bf2f(unsigned short h){
  unsigned int u = ((unsigned int)h) << 16;
  return __builtin_bit_cast(float, u);
}
__device__ __forceinline__ float sigmoidf(float v){ return 1.f/(1.f+__expf(-v)); }

// ---------------------------------------------------------------------------
// Repack conv weights (Cout,Cin,taps) f32 -> bf16 pairs [K/2][Cout] dwords.
// k = tap*Cin + ic ; pair (k, k+1) never crosses a tap (Cin even).
// Dword = {bf16(k even) lo, bf16(k odd) hi} — matches WMMA B 32x16 layout.
// ---------------------------------------------------------------------------
__global__ void k_wprep(const float* __restrict__ src, unsigned int* __restrict__ dst,
                        int Cin, int taps, int Cout){
  int K = Cin * taps;
  int n = (K/2) * Cout;
  int idx = blockIdx.x * 256 + threadIdx.x;
  if (idx >= n) return;
  int kp = idx / Cout, oc = idx % Cout;
  int k0 = kp * 2;
  int tap = k0 / Cin, ic0 = k0 % Cin;
  float f0 = src[((size_t)oc*Cin + ic0    )*taps + tap];
  float f1 = src[((size_t)oc*Cin + ic0 + 1)*taps + tap];
  dst[idx] = (unsigned int)f2bf(f0) | ((unsigned int)f2bf(f1) << 16);
}

// --------------------------- channel-attention stats -----------------------
__global__ void k_colsum(const float* __restrict__ pan, const float* __restrict__ ms,
                         const float* __restrict__ nir, float* __restrict__ out){
  int bc = blockIdx.x;                 // b*192 + c
  int b = bc / 192, c = bc % 192;
  const float* base = (c < 64 ? pan : (c < 128 ? ms : nir)) + ((size_t)b*64 + (c & 63))*HWD;
  float s = 0.f;
  for (int i = threadIdx.x; i < HWD; i += 256) s += base[i];
  __shared__ float red[256];
  red[threadIdx.x] = s; __syncthreads();
  for (int o = 128; o > 0; o >>= 1){
    if (threadIdx.x < o) red[threadIdx.x] += red[threadIdx.x + o];
    __syncthreads();
  }
  if (threadIdx.x == 0) out[bc] = red[0];
}

__global__ void k_camlp(const float* __restrict__ sums, const float* __restrict__ w1,
                        const float* __restrict__ w2, float* __restrict__ ca){
  int b = blockIdx.x, t = threadIdx.x;
  __shared__ float mean[192];
  __shared__ float hid[12];
  if (t < 192) mean[t] = sums[b*192 + t] * (1.0f/(float)HWD);
  __syncthreads();
  if (t < 12){
    float s = 0.f;
    for (int c = 0; c < 192; c++) s += w1[t*192 + c] * mean[c];
    hid[t] = s > 0.f ? s : 0.f;
  }
  __syncthreads();
  if (t < 192){
    float s = 0.f;
    for (int h = 0; h < 12; h++) s += w2[t*12 + h] * hid[h];
    ca[b*192 + t] = sigmoidf(s);
  }
}

// per-pixel mean/max over 192 ca-scaled channels
__global__ void k_avgmax(const float* __restrict__ pan, const float* __restrict__ ms,
                         const float* __restrict__ nir, const float* __restrict__ ca,
                         float* __restrict__ am){
  int i = blockIdx.x*256 + threadIdx.x;
  if (i >= 4*HWD) return;
  int b = i / HWD, p = i % HWD;
  const float* cav = ca + b*192;
  float s = 0.f, mx = -3.4e38f;
  for (int g3 = 0; g3 < 3; g3++){
    const float* base = (g3 == 0 ? pan : (g3 == 1 ? ms : nir));
    for (int c = 0; c < 64; c++){
      float v = base[((size_t)b*64 + c)*HWD + p] * cav[g3*64 + c];
      s += v; mx = v > mx ? v : mx;
    }
  }
  am[((size_t)b*2 + 0)*HWD + p] = s * (1.f/192.f);
  am[((size_t)b*2 + 1)*HWD + p] = mx;
}

__global__ void k_saconv(const float* __restrict__ am, const float* __restrict__ w,
                         float* __restrict__ sa){
  int i = blockIdx.x*256 + threadIdx.x;
  if (i >= 4*HWD) return;
  int b = i / HWD, p = i % HWD;
  int y = p >> 7, x = p & 127;
  float s = 0.f;
  for (int ch = 0; ch < 2; ch++){
    const float* in = am + ((size_t)b*2 + ch)*HWD;
    for (int ky = 0; ky < 7; ky++){
      int yy = y + ky - 3;
      if (yy < 0 || yy >= 128) continue;
      for (int kx = 0; kx < 7; kx++){
        int xx = x + kx - 3;
        if (xx < 0 || xx >= 128) continue;
        s += w[ch*49 + ky*7 + kx] * in[yy*128 + xx];
      }
    }
  }
  sa[i] = sigmoidf(s);
}

// g = combined * (ca*sa + 1)  -> bf16, layout (B,192,H,W)
__global__ void k_buildg(const float* __restrict__ pan, const float* __restrict__ ms,
                         const float* __restrict__ nir, const float* __restrict__ ca,
                         const float* __restrict__ sa, unsigned short* __restrict__ g){
  size_t i = (size_t)blockIdx.x*256 + threadIdx.x;
  if (i >= (size_t)4*192*HWD) return;
  int p = (int)(i % HWD);
  int c = (int)((i / HWD) % 192);
  int b = (int)(i / ((size_t)192*HWD));
  const float* base = (c < 64 ? pan : (c < 128 ? ms : nir));
  float v = base[((size_t)b*64 + (c & 63))*HWD + p];
  float scale = ca[b*192 + c] * sa[(size_t)b*HWD + p] + 1.0f;
  g[i] = f2bf(v * scale);
}

// ---------------------------------------------------------------------------
// Implicit-GEMM conv via v_wmma_f32_16x16x32_bf16.
// Block = 256 threads (8 wave32), one 32-pixel row tile; Cout = NPASS*64.
// K = taps*Cin (multiple of 192), staged to LDS in 192-K chunks as [m][k]+2 pad.
// wave w: m_sub = w&1 (16 rows), n_sub = w>>1 (16 cols), NPASS col groups.
// Next chunk's weight sub-tile is prefetched (global_prefetch_b8) while the
// current chunk's 6*NPASS WMMAs per wave execute.
// ---------------------------------------------------------------------------
template<int NPASS>
__global__ __launch_bounds__(256)
void k_gemm(const unsigned short* __restrict__ X, const unsigned int* __restrict__ Wq,
            const float* __restrict__ bias, const float* __restrict__ residual,
            void* __restrict__ Out, int Cin, int taps, int act)
{
  const int Cout = NPASS * 64;
  const int Ktot = Cin * taps;
  const int H = 128, Wd = 128;

  int tile = blockIdx.x;
  int tx = tile & 3;            // W/32 = 4 tiles per row
  int y  = (tile >> 2) & 127;
  int b  = tile >> 9;
  int x0 = tx * 32;

  int tid  = threadIdx.x;
  int lane = tid & 31, wid = tid >> 5;
  int m_sub = wid & 1, n_sub = wid >> 1;
  int half  = lane >> 4;
  int mloc  = m_sub*16 + (lane & 15);
  int ncol  = lane & 15;

  __shared__ unsigned short Alds[32 * 194];   // 32 pixels x (192+2 pad) bf16

  const v8f vzero = {0.f,0.f,0.f,0.f,0.f,0.f,0.f,0.f};
  v8f acc[NPASS];
  #pragma unroll
  for (int p = 0; p < NPASS; p++) acc[p] = vzero;

  for (int k0 = 0; k0 < Ktot; k0 += 192){
    __syncthreads();
    {   // stage A chunk: Alds[m][kl], kl in [0,192)
      int m = tid & 31;
      for (int kl = tid >> 5; kl < 192; kl += 8){
        int kk  = k0 + kl;
        int ic  = kk % Cin;
        int tap = kk / Cin;
        int dy = (taps == 9) ? (tap/3 - 1) : 0;
        int dx = (taps == 9) ? (tap%3 - 1) : 0;
        int yy = y + dy, xx = x0 + m + dx;
        unsigned short v = 0;
        if (yy >= 0 && yy < H && xx >= 0 && xx < Wd)
          v = X[(((size_t)b*Cin + ic)*H + yy)*Wd + xx];
        Alds[m*194 + kl] = v;
      }
    }
    __syncthreads();

    // prefetch next chunk's weight sub-tile (96 dword rows x Cout) into
    // near caches while this chunk's WMMAs run: global_prefetch_b8
    if (k0 + 192 < Ktot){
      size_t nbase = (size_t)((k0 + 192)/2) * Cout;     // dwords
      size_t ndw   = (size_t)96 * Cout;                 // dwords in sub-tile
      for (size_t ofs = (size_t)tid*32; ofs < ndw; ofs += (size_t)256*32)
        __builtin_prefetch(&Wq[nbase + ofs], 0, 1);
    }

    #pragma unroll
    for (int ks = 0; ks < 192; ks += 32){
      // A fragment: 8 packed bf16 pairs per lane (contiguous k = 16*half + s)
      v8u au;
      #pragma unroll
      for (int v = 0; v < 8; v++){
        int kl = ks + half*16 + v*2;
        au[v] = *(const unsigned int*)(&Alds[mloc*194 + kl]);
      }
      v16bf afrag = __builtin_bit_cast(v16bf, au);

      #pragma unroll
      for (int p = 0; p < NPASS; p++){
        int nb = p*64 + n_sub*16 + ncol;
        int kprow = (k0 + ks)/2 + half*8;     // dword rows of Wq
        v8u bu;
        #pragma unroll
        for (int v = 0; v < 8; v++)
          bu[v] = Wq[(size_t)(kprow + v)*Cout + nb];
        v16bf bfrag = __builtin_bit_cast(v16bf, bu);
        acc[p] = __builtin_amdgcn_wmma_f32_16x16x32_bf16(
            false, afrag, false, bfrag, (short)0, acc[p], false, false);
      }
    }
  }

  // epilogue: C/D layout — VGPR v: M = half*8 + v, N = lane&15
  #pragma unroll
  for (int p = 0; p < NPASS; p++){
    #pragma unroll
    for (int v = 0; v < 8; v++){
      int mrow = half*8 + v;
      int oc   = p*64 + n_sub*16 + ncol;
      int m    = m_sub*16 + mrow;
      size_t idx = (((size_t)b*Cout + oc)*H + y)*Wd + (x0 + m);
      float val = acc[p][v] + bias[oc];
      if (residual){
        ((float*)Out)[idx] = val + residual[idx];
      } else {
        if (act == 1)      val = val > 0.f ? val : 0.f;
        else if (act == 2) val = sigmoidf(val);
        ((unsigned short*)Out)[idx] = f2bf(val);
      }
    }
  }
}

// ---------------------------------------------------------------------------
// Fused dynamic conv + Laplacian high-pass + gating + corr modulation.
// pan 3x3 window loaded once, used for both the 9-tap dynamic conv and lap.
// ---------------------------------------------------------------------------
__global__ void k_dynamic(const unsigned short* __restrict__ dk,
                          const unsigned short* __restrict__ gates,
                          const float* __restrict__ pan, const float* __restrict__ ms,
                          const float* __restrict__ corr, const float* __restrict__ alpha,
                          float* __restrict__ modf, unsigned short* __restrict__ modb)
{
  size_t i = (size_t)blockIdx.x*256 + threadIdx.x;
  if (i >= (size_t)4*64*HWD) return;
  int p = (int)(i % HWD);
  int c = (int)((i / HWD) % 64);
  int b = (int)(i / ((size_t)64*HWD));
  int y = p >> 7, x = p & 127;
  const float* panc = pan + ((size_t)b*64 + c)*HWD;

  float det = 0.f, lap = 0.f;
  #pragma unroll
  for (int t = 0; t < 9; t++){
    int dy = t/3 - 1, dx = t%3 - 1;
    int yy = y + dy, xx = x + dx;
    float pv = (yy >= 0 && yy < 128 && xx >= 0 && xx < 128) ? panc[yy*128 + xx] : 0.f;
    float kv = bf2f(dk[(((size_t)b*576) + c*9 + t)*HWD + p]);
    det += kv * pv;
    lap += ((t == 4) ? 8.f : -1.f) * pv;
  }
  float gp  = bf2f(gates[((size_t)b*128 + c     )*HWD + p]);
  float gm  = bf2f(gates[((size_t)b*128 + 64 + c)*HWD + p]);
  float msv = ms[((size_t)b*64 + c)*HWD + p];
  float fused = gp*det + gm*msv + alpha[0]*lap;
  float m = fused * corr[(size_t)b*HWD + p];
  modf[i] = m;
  modb[i] = f2bf(m);
}

// ---------------------------------------------------------------------------
extern "C" void kernel_launch(void* const* d_in, const int* in_sizes, int n_in,
                              void* d_out, int out_size, void* d_ws, size_t ws_size,
                              hipStream_t stream)
{
  (void)in_sizes; (void)n_in; (void)out_size; (void)ws_size;
  const float* pan    = (const float*)d_in[0];
  const float* ms     = (const float*)d_in[1];
  const float* nir    = (const float*)d_in[2];
  const float* corr   = (const float*)d_in[3];
  const float* ca_w1  = (const float*)d_in[4];
  const float* ca_w2  = (const float*)d_in[5];
  const float* sa_w   = (const float*)d_in[6];
  const float* gate_w = (const float*)d_in[7];
  const float* gate_b = (const float*)d_in[8];
  const float* kp_w1  = (const float*)d_in[9];
  const float* kp_b1  = (const float*)d_in[10];
  const float* kp_w2  = (const float*)d_in[11];
  const float* kp_b2  = (const float*)d_in[12];
  const float* alpha  = (const float*)d_in[13];
  const float* fc_w   = (const float*)d_in[14];
  const float* fc_b   = (const float*)d_in[15];

  char* ws = (char*)d_ws; size_t off = 0;
  auto alloc = [&](size_t bytes)->char*{
    char* p = ws + off; off = (off + bytes + 255) & ~(size_t)255; return p;
  };
  unsigned int*   w1q   = (unsigned int*)  alloc(1728u*192u*2u);   // kp_w1 packed
  unsigned int*   w2q   = (unsigned int*)  alloc(192u*576u*2u);    // kp_w2 packed
  unsigned int*   gwq   = (unsigned int*)  alloc(192u*128u*2u);    // gate_w packed
  unsigned int*   fcq   = (unsigned int*)  alloc(576u*64u*2u);     // fc_w packed
  float*          casum = (float*)         alloc(4u*192u*4u);
  float*          ca    = (float*)         alloc(4u*192u*4u);
  float*          am    = (float*)         alloc((size_t)4*2*HWD*4);
  float*          sa    = (float*)         alloc((size_t)4*HWD*4);
  unsigned short* g     = (unsigned short*)alloc((size_t)4*192*HWD*2);
  unsigned short* hid   = (unsigned short*)alloc((size_t)4*192*HWD*2);
  unsigned short* dk    = (unsigned short*)alloc((size_t)4*576*HWD*2);
  unsigned short* gates = (unsigned short*)alloc((size_t)4*128*HWD*2);
  float*          modf  = (float*)         alloc((size_t)4*64*HWD*4);
  unsigned short* modb  = (unsigned short*)alloc((size_t)4*64*HWD*2);

  // weight repack
  k_wprep<<<(1728/2*192 + 255)/256, 256, 0, stream>>>(kp_w1,  w1q, 192, 9, 192);
  k_wprep<<<(192/2*576  + 255)/256, 256, 0, stream>>>(kp_w2,  w2q, 192, 1, 576);
  k_wprep<<<(192/2*128  + 255)/256, 256, 0, stream>>>(gate_w, gwq, 192, 1, 128);
  k_wprep<<<(576/2*64   + 255)/256, 256, 0, stream>>>(fc_w,   fcq,  64, 9,  64);

  // attention chain
  k_colsum<<<4*192, 256, 0, stream>>>(pan, ms, nir, casum);
  k_camlp <<<4, 256, 0, stream>>>(casum, ca_w1, ca_w2, ca);
  k_avgmax<<<(4*HWD + 255)/256, 256, 0, stream>>>(pan, ms, nir, ca, am);
  k_saconv<<<(4*HWD + 255)/256, 256, 0, stream>>>(am, sa_w, sa);
  k_buildg<<<(int)(((size_t)4*192*HWD + 255)/256), 256, 0, stream>>>(pan, ms, nir, ca, sa, g);

  const int gemmGrid = 4*128*4;   // B * H * (W/32)
  // kp1: 3x3 192->192, ReLU
  k_gemm<3><<<gemmGrid, 256, 0, stream>>>(g,   w1q, kp_b1, nullptr, hid,   192, 9, 1);
  // kp2: 1x1 192->576
  k_gemm<9><<<gemmGrid, 256, 0, stream>>>(hid, w2q, kp_b2, nullptr, dk,    192, 1, 0);
  // gates: 1x1 192->128, sigmoid
  k_gemm<2><<<gemmGrid, 256, 0, stream>>>(g,   gwq, gate_b, nullptr, gates, 192, 1, 2);

  // dynamic conv + laplacian + gating + corr
  k_dynamic<<<(int)(((size_t)4*64*HWD + 255)/256), 256, 0, stream>>>(
      dk, gates, pan, ms, corr, alpha, modf, modb);

  // final: 3x3 64->64 + residual -> f32 d_out
  k_gemm<1><<<gemmGrid, 256, 0, stream>>>(modb, fcq, fc_b, modf, (float*)d_out, 64, 9, 0);
}
